// Block_29729763623137
// MI455X (gfx1250) — matrix operand order repaired and checked by reading, not compiled, not run on hardware
//
#include <hip/hip_runtime.h>
#include <hip/hip_bf16.h>

typedef __bf16 bf16_t;
typedef __attribute__((ext_vector_type(16))) __bf16 v16bf;
typedef __attribute__((ext_vector_type(8)))  __bf16 v8bf;
typedef __attribute__((ext_vector_type(8)))  float  v8f;

constexpr int Bc = 2, Tc = 2048, Dc = 1024, Hc = 16, HDc = 64, Fc = 4096;
constexpr int Mrows = Bc * Tc;          // 4096 token rows
constexpr float LN_EPS = 1e-5f;
// reference scales by D^-0.5 = 1/32; fold log2(e) in for exp2-based softmax.
// This factor is pre-applied to Q in the QKV projection epilogue.
constexpr float ATT_SCALE = 0.03125f * 1.44269504088896340736f;

__device__ __forceinline__ v16bf cat8(v8bf lo, v8bf hi) {
  return __builtin_shufflevector(lo, hi, 0,1,2,3,4,5,6,7,8,9,10,11,12,13,14,15);
}
// A-fragment (16-bit A 16x32): per lane elements K = kh*8+j (j=0..7) then K = 16+kh*8+j
__device__ __forceinline__ v16bf ldfragA(const bf16_t* p) {
  v8bf lo = *(const v8bf*)p;
  v8bf hi = *(const v8bf*)(p + 16);
  return cat8(lo, hi);
}
// B-fragment (16-bit B 32x16): per lane elements K = kh*8+j (j=0..7) then K = 16+kh*8+j
__device__ __forceinline__ v16bf ldfragB(const bf16_t* p) {
  v8bf lo = *(const v8bf*)p;
  v8bf hi = *(const v8bf*)(p + 16);
  return cat8(lo, hi);
}
__device__ __forceinline__ v8f wmma_bf16(v16bf a, v16bf b, v8f c) {
  return __builtin_amdgcn_wmma_f32_16x16x32_bf16(
      /*neg_a=*/false, a, /*neg_b=*/false, b,
      /*c_mod=*/(short)0, c, /*reuse_a=*/false, /*reuse_b=*/false);
}
__device__ __forceinline__ float hmax16(float v) {
#pragma unroll
  for (int m = 1; m < 16; m <<= 1) v = fmaxf(v, __shfl_xor(v, m, 32));
  return v;
}
__device__ __forceinline__ float hsum16(float v) {
#pragma unroll
  for (int m = 1; m < 16; m <<= 1) v += __shfl_xor(v, m, 32);
  return v;
}

// ---------------- LayerNorm: f32 row -> bf16 row ----------------
__global__ __launch_bounds__(256) void ln_bf16_kernel(
    const float* __restrict__ x, const float* __restrict__ g,
    const float* __restrict__ bb, bf16_t* __restrict__ out) {
  const int row = blockIdx.x;
  const int tid = threadIdx.x;
  const float4* xr = (const float4*)(x + (size_t)row * Dc);
  float4 v = xr[tid];
  float s  = v.x + v.y + v.z + v.w;
  float q  = v.x * v.x + v.y * v.y + v.z * v.z + v.w * v.w;
  __shared__ float ss[256], sq[256];
  ss[tid] = s; sq[tid] = q;
  __syncthreads();
  for (int st = 128; st > 0; st >>= 1) {
    if (tid < st) { ss[tid] += ss[tid + st]; sq[tid] += sq[tid + st]; }
    __syncthreads();
  }
  const float mu   = ss[0] * (1.0f / Dc);
  const float var  = sq[0] * (1.0f / Dc) - mu * mu;
  const float rstd = rsqrtf(var + LN_EPS);
  float4 gv = ((const float4*)g)[tid];
  float4 bv = ((const float4*)bb)[tid];
  bf16_t* orow = out + (size_t)row * Dc + tid * 4;
  orow[0] = (bf16_t)((v.x - mu) * rstd * gv.x + bv.x);
  orow[1] = (bf16_t)((v.y - mu) * rstd * gv.y + bv.y);
  orow[2] = (bf16_t)((v.z - mu) * rstd * gv.z + bv.z);
  orow[3] = (bf16_t)((v.w - mu) * rstd * gv.w + bv.w);
}

// --------- weight transpose+convert: [K,N] f32 -> [N,K] bf16 ---------
__global__ __launch_bounds__(256) void transpose_bf16_kernel(
    const float* __restrict__ src, bf16_t* __restrict__ dst, int K, int N) {
  size_t idx = (size_t)blockIdx.x * 256 + threadIdx.x;
  if (idx >= (size_t)K * N) return;
  int n = (int)(idx / K);
  int k = (int)(idx % K);
  dst[idx] = (bf16_t)src[(size_t)k * N + n];
}
// [H, K, HD] f32 -> [(h*HD+e), K] bf16 (flattened per-head proj as one [N=D,K=D] GEMM)
__global__ __launch_bounds__(256) void transpose_head_kernel(
    const float* __restrict__ src, bf16_t* __restrict__ dst) {
  size_t idx = (size_t)blockIdx.x * 256 + threadIdx.x; // over 1024*1024
  int n = (int)(idx >> 10);          // h*HD + e
  int k = (int)(idx & 1023);         // d
  int h = n >> 6, e = n & 63;
  dst[idx] = (bf16_t)src[((size_t)h * Dc + k) * HDc + e];
}

// ---------------- generic bf16 WMMA GEMM (64x64 tile / wave) ----------------
// block = 8 waves as 2(M) x 4(N) -> 128x256 block tile; grid (N/256, M/128)
enum { MODE_BF16 = 0, MODE_VT = 1, MODE_RESID = 2, MODE_RELU = 3 };

template <int MODE>
__global__ __launch_bounds__(256) void gemm_bf16_kernel(
    const bf16_t* __restrict__ A,   // [M,K] bf16 row-major
    const bf16_t* __restrict__ Wt,  // [N,K] bf16 row-major (pre-transposed weights)
    const float* __restrict__ bias, const float* __restrict__ resid,
    void* __restrict__ outv, int M, int N, int K, float oscale) {
  const int lane = threadIdx.x & 31;
  const int wave = threadIdx.x >> 5;
  const int tileM = blockIdx.y * 128 + (wave & 1) * 64;
  const int tileN = blockIdx.x * 256 + (wave >> 1) * 64;
  const int r0 = lane & 15;     // A row / B col within 16-wide sub-tile
  const int kh = lane >> 4;     // half-wave selector
  const bf16_t* ap = A  + (size_t)(tileM + r0) * K + kh * 8;
  const bf16_t* bp = Wt + (size_t)(tileN + r0) * K + kh * 8;
  const size_t ld16 = (size_t)16 * K;

  v8f acc[4][4] = {};
  for (int kk = 0; kk < K; kk += 32) {
    __builtin_prefetch(ap + kk + 256, 0, 0);
    v16bf a[4], b[4];
#pragma unroll
    for (int i = 0; i < 4; ++i) {
      a[i] = ldfragA(ap + i * ld16 + kk);
      b[i] = ldfragB(bp + i * ld16 + kk);
    }
#pragma unroll
    for (int mi = 0; mi < 4; ++mi)
#pragma unroll
      for (int ni = 0; ni < 4; ++ni)
        acc[mi][ni] = wmma_bf16(a[mi], b[ni], acc[mi][ni]);
  }
#pragma unroll
  for (int mi = 0; mi < 4; ++mi) {
    const int rbase = tileM + mi * 16 + (kh << 3);
#pragma unroll
    for (int ni = 0; ni < 4; ++ni) {
      const int col = tileN + ni * 16 + r0;
#pragma unroll
      for (int r = 0; r < 8; ++r) {
        const int row = rbase + r;
        float v = acc[mi][ni][r];
        if (MODE == MODE_BF16) {
          ((bf16_t*)outv)[(size_t)row * N + col] = (bf16_t)(v * oscale);
        } else if (MODE == MODE_VT) {
          // store V transposed: [B, H, HD, T]
          int bidx = row / Tc, t = row % Tc;
          ((bf16_t*)outv)[(((size_t)bidx * Hc + (col >> 6)) * HDc + (col & 63)) * Tc + t] =
              (bf16_t)v;
        } else if (MODE == MODE_RESID) {
          ((float*)outv)[(size_t)row * N + col] =
              resid[(size_t)row * N + col] + v + bias[col];
        } else { // MODE_RELU
          float z = v + bias[col];
          ((bf16_t*)outv)[(size_t)row * N + col] = (bf16_t)(z > 0.0f ? z : 0.0f);
        }
      }
    }
  }
}

// ---------------- causal flash attention (bf16 WMMA) ----------------
// grid: (T/64, H, B), block: 128 (4 waves); each wave: 16 query rows.
// Q is pre-scaled by ATT_SCALE, so S fragments feed exp2 directly.
__global__ __launch_bounds__(128) void attn_kernel(
    const bf16_t* __restrict__ q,   // [B,T,H,HD] == [M, D], pre-scaled
    const bf16_t* __restrict__ k,   // [B,T,H,HD]
    const bf16_t* __restrict__ vT,  // [B,H,HD,T]
    bf16_t* __restrict__ o) {       // [B,T,H,HD]
  __shared__ __align__(16) bf16_t pbuf[4][16][32];
  const int lane = threadIdx.x & 31;
  const int wave = threadIdx.x >> 5;
  const int b = blockIdx.z, h = blockIdx.y;
  const int qbase = blockIdx.x * 64 + wave * 16;
  const int r0 = lane & 15, kh = lane >> 4;

  const bf16_t* qb = q  + ((size_t)b * Tc) * Dc + h * 64;
  const bf16_t* kb = k  + ((size_t)b * Tc) * Dc + h * 64;
  const bf16_t* vb = vT + (((size_t)b * Hc + h) * HDc) * Tc;

  // preload Q fragments for K-dim 0..31 and 32..63
  const bf16_t* qp = qb + (size_t)(qbase + r0) * Dc + kh * 8;
  v16bf qa0 = ldfragA(qp);
  v16bf qa1 = ldfragA(qp + 32);

  float m_run[8], l_run[8];
#pragma unroll
  for (int r = 0; r < 8; ++r) { m_run[r] = -3.0e38f; l_run[r] = 0.0f; }
  v8f accO[4] = {};

  for (int kv = 0; kv < qbase + 16; kv += 32) {
    // ---- S = Q @ K^T for 32 keys (two 16-col fragments) ----
    v8f s0 = {}, s1 = {};
    {
      const bf16_t* kp0 = kb + (size_t)(kv + r0) * Dc + kh * 8;        // cols kv..kv+15
      const bf16_t* kp1 = kb + (size_t)(kv + 16 + r0) * Dc + kh * 8;   // cols kv+16..kv+31
      s0 = wmma_bf16(qa0, ldfragB(kp0), s0);
      s0 = wmma_bf16(qa1, ldfragB(kp0 + 32), s0);
      s1 = wmma_bf16(qa0, ldfragB(kp1), s1);
      s1 = wmma_bf16(qa1, ldfragB(kp1 + 32), s1);
    }
    const int col0 = kv + r0, col1 = col0 + 16;
    const int rowb = qbase + (kh << 3);
    const bool full = (kv + 31 <= qbase);   // wave-uniform: tile fully unmasked
#pragma unroll
    for (int r = 0; r < 8; ++r) {
      float x0, x1;
      if (full) {
        x0 = s0[r];
        x1 = s1[r];
      } else {
        const int row = rowb + r;
        x0 = (col0 <= row) ? s0[r] : -3.0e38f;
        x1 = (col1 <= row) ? s1[r] : -3.0e38f;
      }
      float tmax = hmax16(fmaxf(x0, x1));
      float m_new = fmaxf(m_run[r], tmax);
      float alpha = exp2f(m_run[r] - m_new);
      float p0 = exp2f(x0 - m_new);
      float p1 = exp2f(x1 - m_new);
      l_run[r] = l_run[r] * alpha + hsum16(p0 + p1);
      m_run[r] = m_new;
#pragma unroll
      for (int f = 0; f < 4; ++f) accO[f][r] = accO[f][r] * alpha;
      pbuf[wave][r + (kh << 3)][r0]      = (bf16_t)p0;
      pbuf[wave][r + (kh << 3)][r0 + 16] = (bf16_t)p1;
    }
    __builtin_amdgcn_wave_barrier();   // intra-wave LDS write -> read ordering point
    v16bf pa = ldfragA(&pbuf[wave][r0][kh * 8]);
    // ---- O += P @ V  (N=64 as 4 fragments) ----
#pragma unroll
    for (int f = 0; f < 4; ++f) {
      const bf16_t* vp = vb + (size_t)(16 * f + r0) * Tc + kv + kh * 8;
      accO[f] = wmma_bf16(pa, ldfragB(vp), accO[f]);
    }
    __builtin_amdgcn_wave_barrier();
  }

  // ---- normalize and store O (bf16, [B,T,H,HD]) ----
#pragma unroll
  for (int r = 0; r < 8; ++r) {
    const int row = qbase + r + (kh << 3);
    const float inv = 1.0f / l_run[r];
    bf16_t* op = o + ((size_t)b * Tc + row) * Dc + h * 64 + r0;
#pragma unroll
    for (int f = 0; f < 4; ++f) op[16 * f] = (bf16_t)(accO[f][r] * inv);
  }
}

// ---------------------------------------------------------------
extern "C" void kernel_launch(void* const* d_in, const int* in_sizes, int n_in,
                              void* d_out, int out_size, void* d_ws, size_t ws_size,
                              hipStream_t stream) {
  (void)in_sizes; (void)n_in; (void)out_size; (void)ws_size;
  const float* x    = (const float*)d_in[0];
  const float* Wq   = (const float*)d_in[1];
  const float* Wk   = (const float*)d_in[2];
  const float* Wv   = (const float*)d_in[3];
  const float* Wo   = (const float*)d_in[4];
  const float* bo   = (const float*)d_in[5];
  const float* W1   = (const float*)d_in[6];
  const float* b1   = (const float*)d_in[7];
  const float* W2   = (const float*)d_in[8];
  const float* b2   = (const float*)d_in[9];
  const float* ln1g = (const float*)d_in[10];
  const float* ln1b = (const float*)d_in[11];
  const float* ln2g = (const float*)d_in[12];
  const float* ln2b = (const float*)d_in[13];

  char* base = (char*)d_ws;
  size_t off = 0;
  auto alloc = [&](size_t bytes) -> void* {
    void* p = base + off;
    off += (bytes + 255) & ~(size_t)255;
    return p;
  };
  const size_t MD2 = (size_t)Mrows * Dc * 2;   // bf16 [4096,1024]
  bf16_t* hbf1 = (bf16_t*)alloc(MD2);
  bf16_t* hbf2 = (bf16_t*)alloc(MD2);
  bf16_t* Wqt  = (bf16_t*)alloc((size_t)Dc * Dc * 2);
  bf16_t* Wkt  = (bf16_t*)alloc((size_t)Dc * Dc * 2);
  bf16_t* Wvt  = (bf16_t*)alloc((size_t)Dc * Dc * 2);
  bf16_t* Wot  = (bf16_t*)alloc((size_t)Dc * Dc * 2);
  bf16_t* W1t  = (bf16_t*)alloc((size_t)Fc * Dc * 2);
  bf16_t* W2t  = (bf16_t*)alloc((size_t)Dc * Fc * 2);
  bf16_t* qbf  = (bf16_t*)alloc(MD2);
  bf16_t* kbf  = (bf16_t*)alloc(MD2);
  bf16_t* vTt  = (bf16_t*)alloc(MD2);
  bf16_t* obf  = (bf16_t*)alloc(MD2);
  float*  x2   = (float*)alloc((size_t)Mrows * Dc * 4);
  bf16_t* a1   = (bf16_t*)alloc((size_t)Mrows * Fc * 2);

  // 1) LN1 -> bf16
  ln_bf16_kernel<<<Mrows, 256, 0, stream>>>(x, ln1g, ln1b, hbf1);
  // 2) weight convert/transpose to [N,K] bf16
  transpose_head_kernel<<<(Dc * Dc) / 256, 256, 0, stream>>>(Wq, Wqt);
  transpose_head_kernel<<<(Dc * Dc) / 256, 256, 0, stream>>>(Wk, Wkt);
  transpose_head_kernel<<<(Dc * Dc) / 256, 256, 0, stream>>>(Wv, Wvt);
  transpose_bf16_kernel<<<(Dc * Dc) / 256, 256, 0, stream>>>(Wo, Wot, Dc, Dc);
  transpose_bf16_kernel<<<(Dc * Fc) / 256, 256, 0, stream>>>(W1, W1t, Dc, Fc);
  transpose_bf16_kernel<<<(Fc * Dc) / 256, 256, 0, stream>>>(W2, W2t, Fc, Dc);
  // 3) QKV projections (Q pre-scaled for softmax)
  dim3 gD(Dc / 256, Mrows / 128);
  gemm_bf16_kernel<MODE_BF16><<<gD, 256, 0, stream>>>(hbf1, Wqt, nullptr, nullptr, qbf, Mrows, Dc, Dc, ATT_SCALE);
  gemm_bf16_kernel<MODE_BF16><<<gD, 256, 0, stream>>>(hbf1, Wkt, nullptr, nullptr, kbf, Mrows, Dc, Dc, 1.0f);
  gemm_bf16_kernel<MODE_VT  ><<<gD, 256, 0, stream>>>(hbf1, Wvt, nullptr, nullptr, vTt, Mrows, Dc, Dc, 1.0f);
  // 4) causal attention
  attn_kernel<<<dim3(Tc / 64, Hc, Bc), 128, 0, stream>>>(qbf, kbf, vTt, obf);
  // 5) output projection + residual -> x2 (f32)
  gemm_bf16_kernel<MODE_RESID><<<gD, 256, 0, stream>>>(obf, Wot, bo, x, x2, Mrows, Dc, Dc, 1.0f);
  // 6) LN2 -> bf16
  ln_bf16_kernel<<<Mrows, 256, 0, stream>>>(x2, ln2g, ln2b, hbf2);
  // 7) FFN up + ReLU -> bf16
  gemm_bf16_kernel<MODE_RELU><<<dim3(Fc / 256, Mrows / 128), 256, 0, stream>>>(
      hbf2, W1t, b1, nullptr, a1, Mrows, Fc, Dc, 1.0f);
  // 8) FFN down + residual -> d_out (f32)
  gemm_bf16_kernel<MODE_RESID><<<gD, 256, 0, stream>>>(
      a1, W2t, b2, x2, (float*)d_out, Mrows, Dc, Fc, 1.0f);
}
